// GMM_18253611008351
// MI455X (gfx1250) — matrix operand order repaired and checked by reading, not compiled, not run on hardware
//
#include <hip/hip_runtime.h>
#include <hip/hip_bf16.h>

// ---------------------------------------------------------------------------
// GMM responsibilities on MI455X (gfx1250, wave32)
//   N = 1,000,000 rows, D = 32 dims, K = 8 components
//
// Phase 1 (gmm_prep): 8 blocks x 1 wave. Gauss-Jordan inversion of each 32x32
//   sigma with partial pivoting + log|det|. Expansion form:
//     (x-mu)' B (x-mu) = x'Bx + w'x + c,  w = -2*B*mu, c = mu'B*mu
//   Emits: B pre-packed as WMMA B-operand (32x16 bf16 lane layout),
//          w table (f32), and per-k coef / bias' (c and exp(log_den) folded).
//
// Phase 2 (gmm_main): one wave per 16-row tile.
//   A = bf16(X tile), packed ONCE, reused for all k (A-layout 16x32).
//   Per k: Y = A @ B_k           -> 2x v_wmma_f32_16x16x32_bf16
//          q[n] = sum_d X[n,d]*(Y[n,d] + w_d)   (w'x folded, f32)
//          reduce via DPP butterfly (quad_perm/row_half_mirror/row_mirror)
//          aff = coef_k*q + bias'_k
//   Row softmax over K=8 fused via per-wave LDS transpose; float4 stores.
//   B operands staged to LDS with global_load_async_to_lds_b128 (ASYNCcnt).
// ---------------------------------------------------------------------------

typedef __bf16 v16bf __attribute__((ext_vector_type(16)));
typedef float  v8f   __attribute__((ext_vector_type(8)));

#define GMM_D 32
#define GMM_K 8

// DPP butterfly add over the 16 lanes of each half-wave group.
#define DPP_ADD(x, ctrl)                                                      \
  ((x) + __int_as_float(__builtin_amdgcn_update_dpp(                          \
             0, __float_as_int(x), (ctrl), 0xf, 0xf, true)))

__device__ __forceinline__ float reduce16(float q) {
  q = DPP_ADD(q, 0xB1);   // quad_perm [1,0,3,2]  (xor 1)
  q = DPP_ADD(q, 0x4E);   // quad_perm [2,3,0,1]  (xor 2)
  q = DPP_ADD(q, 0x141);  // row_half_mirror      (xor 4 on quad-uniform data)
  q = DPP_ADD(q, 0x140);  // row_mirror           (xor 8 on 8-uniform data)
  return q;
}

// ---------------------------------------------------------------------------
// Phase 1: per-k 32x32 inversion + slogdet + parameter folding
// grid = K blocks, block = 32 threads (one wave), lane = row/column index
// ---------------------------------------------------------------------------
__global__ void gmm_prep(const float* __restrict__ sigmas,
                         const float* __restrict__ mus,
                         const float* __restrict__ phis,
                         __bf16* __restrict__ wsB,
                         float* __restrict__ wsP,
                         float* __restrict__ wsW) {
  __shared__ float aug[32 * 64];  // [row][64]: [A | I]
  const int k = blockIdx.x;
  const int l = threadIdx.x;  // 0..31

  for (int i = 0; i < 32; ++i) {
    aug[i * 64 + l]      = sigmas[k * 1024 + i * 32 + l];
    aug[i * 64 + 32 + l] = (i == l) ? 1.0f : 0.0f;
  }
  __syncthreads();

  float logdet = 0.0f;
  for (int p = 0; p < 32; ++p) {
    // partial-pivot argmax over rows p..31 (lane = row candidate)
    float bv = (l >= p) ? fabsf(aug[l * 64 + p]) : -1.0f;
    int   bi = l;
#pragma unroll
    for (int off = 16; off > 0; off >>= 1) {
      float ov = __shfl_xor(bv, off);
      int   oi = __shfl_xor(bi, off);
      if (ov > bv || (ov == bv && oi < bi)) { bv = ov; bi = oi; }
    }
    const int pr = bi;
    __syncthreads();
    if (pr != p) {  // lane = column: swap rows p <-> pr (cols l and l+32)
      float t0 = aug[p * 64 + l];
      aug[p * 64 + l] = aug[pr * 64 + l];
      aug[pr * 64 + l] = t0;
      float t1 = aug[p * 64 + 32 + l];
      aug[p * 64 + 32 + l] = aug[pr * 64 + 32 + l];
      aug[pr * 64 + 32 + l] = t1;
    }
    __syncthreads();
    const float piv = aug[p * 64 + p];
    logdet += logf(fabsf(piv));
    const float ip = 1.0f / piv;
    aug[p * 64 + l]      *= ip;
    aug[p * 64 + 32 + l] *= ip;
    __syncthreads();
    // lane = row: eliminate column p from all rows != p
    const float f = aug[l * 64 + p];
    if (l != p) {
      for (int c = 0; c < 64; ++c) aug[l * 64 + c] -= f * aug[p * 64 + c];
    }
    __syncthreads();
  }

  // Pack Sigma^-1 as WMMA B operand (32x16 bf16, wave32), contiguous k order (same as A):
  //   lane: N-col = l%16, K-rows = 16*(l/16) + {0..15}; 16 bf16 contiguous.
  for (int h = 0; h < 2; ++h) {
    const int d     = 16 * h + (l & 15);
    const int ebase = 16 * (l >> 4);
#pragma unroll
    for (int t = 0; t < 16; ++t) {
      wsB[((k * 2 + h) * 32 + l) * 16 + t] = (__bf16)aug[(ebase + t) * 64 + 32 + d];
    }
  }

  // acc_d = sum_e mu_e * Inv[e][d]   (lane = d)
  float acc = 0.0f;
  for (int e = 0; e < 32; ++e) acc += mus[k * 32 + e] * aug[e * 64 + 32 + l];
  wsW[k * 32 + l] = -2.0f * acc;  // w = -2 * mu' * B  (row vector)

  // cK = mu' B mu = sum_d mu_d * acc_d  (32-lane butterfly)
  float tq = mus[k * 32 + l] * acc;
#pragma unroll
  for (int off = 16; off > 0; off >>= 1) tq += __shfl_xor(tq, off);

  if (l == 0) {
    const float phi  = phis[k];
    const float coef = -0.5f * phi;
    const float log_den = 0.5f * (32.0f * 1.8378770664093454f + logdet);
    wsP[k]     = coef;
    wsP[8 + k] = coef * tq - phi * __expf(log_den);  // coef*c + bias
  }
}

// ---------------------------------------------------------------------------
// Phase 2: main pass. block = 128 threads (4 waves), 16 rows per wave.
// ---------------------------------------------------------------------------
__global__ void __launch_bounds__(128, 4)
gmm_main(const float* __restrict__ X,
         const __bf16* __restrict__ wsB,
         const float* __restrict__ wsP,
         const float* __restrict__ wsW,
         float* __restrict__ out, int N) {
  __shared__ __align__(32) __bf16 sB[GMM_K * 2 * 32 * 16];  // 16 KB B operands
  __shared__ __align__(16) float sX[4 * 16 * GMM_D];        // 8 KB: X tiles
  __shared__ __align__(16) float sW[GMM_K * GMM_D];         // 1 KB: w vectors
  __shared__ float sParam[16];                              // coef[8], bias'[8]
  __shared__ float sAff[4 * 16 * GMM_K];                    // per-wave transpose

  const int t = threadIdx.x;
  const int wave = t >> 5;
  const int lane = t & 31;
  const long tile    = (long)blockIdx.x * 4 + wave;
  const long rowbase = tile * 16;

  // --- Async-stage B operands (16 KB) into LDS: 128 threads x 8 x b128. ---
  {
    // Low 32 bits of a generic pointer to LDS == LDS address (aperture rule).
    const unsigned ldsBase = (unsigned)(unsigned long long)(const void*)sB;
#pragma unroll
    for (int i = 0; i < 8; ++i) {
      const unsigned ldsOff = ldsBase + (unsigned)(t + i * 128) * 16u;
      const char*    gsrc   = (const char*)wsB + (size_t)(t + i * 128) * 16u;
      asm volatile("global_load_async_to_lds_b128 %0, %1, off"
                   :
                   : "v"(ldsOff), "v"(gsrc)
                   : "memory");
    }
  }

  // --- Coalesced X tile load -> LDS (one wave's 16x32 tile = 2 KB). ---
  if (rowbase < N) {
    float* sXw = sX + wave * 16 * GMM_D;
#pragma unroll
    for (int i = 0; i < 4; ++i) {
      long e = rowbase * GMM_D + (long)(lane + 32 * i) * 4;  // flat f32 index
      if (e + 4 > (long)N * GMM_D) e = (long)N * GMM_D - 4;  // tail clamp
      const float4 v = *(const float4*)(X + e);
      *(float4*)(sXw + (lane + 32 * i) * 4) = v;
    }
  }

  if (t < 64) ((float4*)sW)[t] = ((const float4*)wsW)[t];
  if (t < 16) sParam[t] = wsP[t];

  asm volatile("s_wait_asynccnt 0x0" ::: "memory");  // B staging complete
  __syncthreads();

  if (rowbase >= N) return;  // wave-uniform; barrier already passed

  const int li = lane & 15;
  const int hi = lane >> 4;
  const float* sXw = sX + wave * 16 * GMM_D;

  // --- A operand: bf16(X) packed ONCE, reused for all 8 components. ---
  // A-layout (16-bit, 16x32): lane = row li; VGPR pair -> K-slice indices.
  v16bf a;
  float xc0[8], xc1[8];
#pragma unroll
  for (int v = 0; v < 8; ++v) {
    const int dA = 16 * hi + 2 * v;
    const float2 xv = *(const float2*)(sXw + li * GMM_D + dA);
    a[2 * v]     = (__bf16)xv.x;
    a[2 * v + 1] = (__bf16)xv.y;
    // X in C/D-matrix order (lane = col li; VGPR v -> row v + 8*hi)
    xc0[v] = sXw[(v + 8 * hi) * GMM_D + li];
    xc1[v] = sXw[(v + 8 * hi) * GMM_D + 16 + li];
  }

  const v8f zero = {0.f, 0.f, 0.f, 0.f, 0.f, 0.f, 0.f, 0.f};

#pragma unroll
  for (int k = 0; k < GMM_K; ++k) {
    const v16bf b0 = *(const v16bf*)(sB + ((k * 2 + 0) * 32 + lane) * 16);
    const v16bf b1 = *(const v16bf*)(sB + ((k * 2 + 1) * 32 + lane) * 16);

    // Y = X @ SigmaInv_k : two 16x16 f32 tiles (d = 0..15, 16..31)
    v8f c0 = __builtin_amdgcn_wmma_f32_16x16x32_bf16(
        false, a, false, b0, (short)0, zero, false, false);
    v8f c1 = __builtin_amdgcn_wmma_f32_16x16x32_bf16(
        false, a, false, b1, (short)0, zero, false, false);

    const float w0 = sW[k * GMM_D + li];        // w_d for this lane's column
    const float w1 = sW[k * GMM_D + 16 + li];
    const float coefk = sParam[k];
    const float biask = sParam[8 + k];

    float aff[8];
#pragma unroll
    for (int v = 0; v < 8; ++v) {
      // q_lane = X[n,d]*(Y[n,d] + w_d) summed over this lane's two columns;
      // the w'x linear term rides along in f32.
      float q = (c0[v] + w0) * xc0[v] + (c1[v] + w1) * xc1[v];
      q = reduce16(q);  // DPP butterfly: full row sum, uniform across 16 lanes
      aff[v] = coefk * q + biask;
    }
    if (li == 0) {  // lanes 0 and 16 publish rows v and v+8 (one exec toggle)
#pragma unroll
      for (int v = 0; v < 8; ++v)
        sAff[wave * 128 + (v + 8 * hi) * GMM_K + k] = aff[v];
    }
  }

  // --- fused row softmax over K=8 (same wave; LDS ops are in-order) ---
  if (lane < 16) {
    const long row = rowbase + lane;
    if (row < N) {
      const float4 v0 = *(const float4*)&sAff[wave * 128 + lane * 8];
      const float4 v1 = *(const float4*)&sAff[wave * 128 + lane * 8 + 4];
      float mx = fmaxf(fmaxf(fmaxf(v0.x, v0.y), fmaxf(v0.z, v0.w)),
                       fmaxf(fmaxf(v1.x, v1.y), fmaxf(v1.z, v1.w)));
      const float e0 = __expf(v0.x - mx), e1 = __expf(v0.y - mx);
      const float e2 = __expf(v0.z - mx), e3 = __expf(v0.w - mx);
      const float e4 = __expf(v1.x - mx), e5 = __expf(v1.y - mx);
      const float e6 = __expf(v1.z - mx), e7 = __expf(v1.w - mx);
      const float is = 1.0f / (e0 + e1 + e2 + e3 + e4 + e5 + e6 + e7);
      float4 o0 = {e0 * is, e1 * is, e2 * is, e3 * is};
      float4 o1 = {e4 * is, e5 * is, e6 * is, e7 * is};
      *(float4*)(out + row * GMM_K)     = o0;
      *(float4*)(out + row * GMM_K + 4) = o1;
    }
  }
}

// ---------------------------------------------------------------------------
extern "C" void kernel_launch(void* const* d_in, const int* in_sizes, int n_in,
                              void* d_out, int out_size, void* d_ws, size_t ws_size,
                              hipStream_t stream) {
  const float* X      = (const float*)d_in[0];
  const float* mus    = (const float*)d_in[1];
  const float* sigmas = (const float*)d_in[2];
  const float* phis   = (const float*)d_in[3];
  float* out = (float*)d_out;

  const int N = in_sizes[0] / GMM_D;

  __bf16* wsB = (__bf16*)d_ws;                  // 16 KB packed B operands
  float*  wsP = (float*)((char*)d_ws + 16384);  // coef[8] + bias'[8]
  float*  wsW = (float*)((char*)d_ws + 16448);  // w[8][32]

  gmm_prep<<<dim3(GMM_K), dim3(32), 0, stream>>>(sigmas, mus, phis,
                                                 wsB, wsP, wsW);

  const int nTiles = (N + 15) / 16;
  const int blocks = (nTiles + 3) / 4;
  gmm_main<<<dim3(blocks), dim3(128), 0, stream>>>(X, wsB, wsP, wsW, out, N);
}
